// MinimalRNN_42726334660623
// MI455X (gfx1250) — matrix operand order repaired and checked by reading, not compiled, not run on hardware
//
#include <hip/hip_runtime.h>
#include <hip/hip_bf16.h>
#include <math.h>
#include <stdint.h>

// ---- problem constants (match reference) ----
constexpr int Hd = 512;    // hidden
constexpr int Bv = 128;    // batch
constexpr int Tv = 2048;   // sequence length
constexpr int NWG = 16;    // workgroups cooperating on the scan (1 per WGP)

// ---- CDNA5 WMMA vector types (wave32) ----
typedef __attribute__((ext_vector_type(16))) __bf16 v16bf;
typedef __attribute__((ext_vector_type(8)))  __bf16 bf16x8;
typedef __attribute__((ext_vector_type(8)))  float  v8f;

// Pointer types matching the async-to-LDS builtin's signature exactly:
// param0: addrspace(1) "vector_size(16) int"*, param1: addrspace(3) same.
typedef int v4i_vs __attribute__((vector_size(16)));
typedef __attribute__((address_space(1))) v4i_vs* g_v4i;
typedef __attribute__((address_space(3))) v4i_vs* l_v4i;

// ---- feature probes ----
#if defined(__has_builtin)
#if __has_builtin(__builtin_amdgcn_global_load_async_to_lds_b128)
#define HAVE_ASYNC_LDS 1
#endif
#if __has_builtin(__builtin_amdgcn_s_wait_asynccnt)
#define HAVE_WAIT_ASYNC 1
#endif
#if __has_builtin(__builtin_amdgcn_s_cluster_barrier)
#define HAVE_CLUSTER_BARRIER 1
#endif
#endif

// 16-byte global->LDS copy: async (ASYNCcnt-tracked, no VGPR round trip) if
// the toolchain exposes it, else plain load+store.
static __device__ inline void copy16_to_lds(void* lds_dst, const void* gsrc) {
#if HAVE_ASYNC_LDS
  __builtin_amdgcn_global_load_async_to_lds_b128(
      (g_v4i)(uintptr_t)gsrc, (l_v4i)(uintptr_t)lds_dst, 0, 0);
#else
  *(bf16x8*)lds_dst = *(const bf16x8*)gsrc;
#endif
}
static __device__ inline void async_copies_done() {
#if HAVE_ASYNC_LDS
#if HAVE_WAIT_ASYNC
  __builtin_amdgcn_s_wait_asynccnt(0);
#else
  asm volatile("s_wait_asynccnt 0" ::: "memory");
#endif
#endif
}

// Build a 16-element bf16 fragment from two contiguous 16-byte halves.
static __device__ inline v16bf load_frag(const __bf16* lo, const __bf16* hi) {
  v16bf a;
  *(bf16x8*)&a       = *(const bf16x8*)lo;   // elements 0..7
  *((bf16x8*)&a + 1) = *(const bf16x8*)hi;   // elements 8..15
  return a;
}

// L2-resident sense barrier across the NWG scan workgroups. Counter is reset
// by the prep kernel each launch, so graph replays stay deterministic.
static __device__ inline void grid_barrier(unsigned* cnt, unsigned target, int tid) {
  __threadfence();
  if (tid == 0) {
    __hip_atomic_fetch_add(cnt, 1u, __ATOMIC_RELEASE, __HIP_MEMORY_SCOPE_AGENT);
    while (__hip_atomic_load(cnt, __ATOMIC_ACQUIRE, __HIP_MEMORY_SCOPE_AGENT) < target) {}
  }
  __syncthreads();
}

// =====================================================================
// Kernel 0 (prep): one-time weight conversion into LDS-image layouts,
// zero h0 planes, reset the scan barrier counter.
//   Wx_img[kc][n][kk] : per-32k-chunk transposed W_x image (bf16, 512KB)
//   Wh_img[n][k]      : fully transposed W_h image         (bf16, 512KB)
// =====================================================================
__global__ __launch_bounds__(256)
void prep(const float* __restrict__ W_fc, __bf16* __restrict__ Wx_img,
          __bf16* __restrict__ Wh_img, __bf16* __restrict__ hbuf,
          unsigned* __restrict__ cnt) {
  const int idx = blockIdx.x * 256 + threadIdx.x;        // 0 .. 262143
  const int k = idx >> 9, n = idx & 511;
  // W_x[k][n] = W_fc[k*H + n]
  Wx_img[(k >> 5) * (Hd * 32) + n * 32 + (k & 31)] = (__bf16)W_fc[idx];
  // W_h[k][n] = W_fc[(H+k)*H + n]  ->  Wh_img[n][k]
  Wh_img[n * Hd + k] = (__bf16)W_fc[Hd * Hd + idx];
  if (idx < 2 * Bv * Hd) hbuf[idx] = (__bf16)0.0f;       // h double-buffer = 0
  if (idx == 0) *cnt = 0u;
}

// =====================================================================
// Kernel 1: xp[t][b][:] = embed[x[b,t]] @ W_x + b_fc        (bf16 WMMA)
// One block = 64 gathered embed rows x all 512 columns (cuts L2 W traffic
// 4x vs 16-row blocks). 8 waves; wave w: n in [64w,64w+64), 4x4 tiles.
// W chunk staging = contiguous 32KB async copy of the pre-built image.
// =====================================================================
__global__ __launch_bounds__(256)
void xp_gemm(const int* __restrict__ x, const float* __restrict__ embed,
             const __bf16* __restrict__ Wx_img, const float* __restrict__ b_fc,
             float* __restrict__ xp) {
  __shared__ __bf16 Alds[64][32];     // A tile  (M=64, K=32)  4KB
  __shared__ __bf16 Wlds[512][32];    // W chunk image [n][kk]  32KB
  __shared__ int    tok[64];

  const int tid = threadIdx.x;
  const int m0  = blockIdx.x * 64;             // global A row base (m = b*T + t)
  if (tid < 64) tok[tid] = x[m0 + tid];
  __syncthreads();

  const int lane = tid & 31;
  const int wv   = tid >> 5;                   // wave 0..7
  const int col  = lane & 15;
  const int hh   = lane >> 4;

  v8f acc[4][4];                               // [mt][nt]
  for (int mt = 0; mt < 4; ++mt)
    for (int nt = 0; nt < 4; ++nt)
      for (int j = 0; j < 8; ++j) acc[mt][nt][j] = 0.0f;

  for (int kc = 0; kc < 16; ++kc) {            // K = 512 in chunks of 32
    const int k0 = kc * 32;
    __syncthreads();                           // LDS reuse guard
    // --- stage A tile: gather embed rows, f32 -> bf16 (1024 float2 / 256 thr)
    for (int i = 0; i < 4; ++i) {
      const int e2 = i * 256 + tid;
      const int r = e2 >> 4;                   // 0..63
      const int c = (e2 & 15) * 2;             // 0..30
      const float2 v2 = *(const float2*)(embed + (long)tok[r] * Hd + k0 + c);
      Alds[r][c]     = (__bf16)v2.x;
      Alds[r][c + 1] = (__bf16)v2.y;
    }
    // --- stage W chunk: contiguous 32KB copy (async-to-LDS path) ---
    {
      const __bf16* wsrc = Wx_img + (long)kc * (Hd * 32);
      __bf16* wdst = &Wlds[0][0];
      for (int i = 0; i < 8; ++i) {
        const int off = (i * 256 + tid) * 8;   // 8 bf16 = 16 bytes per copy
        copy16_to_lds(wdst + off, wsrc + off);
      }
      async_copies_done();
    }
    __syncthreads();
    // --- fragments + 16 WMMAs ---
    v16bf af[4];
    for (int mt = 0; mt < 4; ++mt)
      af[mt] = load_frag(&Alds[mt * 16 + col][hh * 16],
                         &Alds[mt * 16 + col][hh * 16 + 8]);
    for (int nt = 0; nt < 4; ++nt) {
      const int n = wv * 64 + nt * 16 + col;
      const v16bf bfr = load_frag(&Wlds[n][hh * 16], &Wlds[n][hh * 16 + 8]);
      for (int mt = 0; mt < 4; ++mt)
        acc[mt][nt] = __builtin_amdgcn_wmma_f32_16x16x32_bf16(
            false, af[mt], false, bfr, (short)0, acc[mt][nt], false, false);
    }
  }

  // --- epilogue: add bias, store xp in [T][B][H] layout ---
  const int b     = m0 >> 11;                  // m / T   (64 | 2048)
  const int tbase = m0 & (Tv - 1);             // m % T
  for (int nt = 0; nt < 4; ++nt) {
    const int n = wv * 64 + nt * 16 + col;
    const float bias = b_fc[n];
    for (int mt = 0; mt < 4; ++mt)
      for (int j = 0; j < 8; ++j) {
        const int t = tbase + mt * 16 + hh * 8 + j;
        xp[((long)t * Bv + b) * Hd + n] = acc[mt][nt][j] + bias;
      }
  }
}

// =====================================================================
// Kernel 2: the scan, spread over NWG=16 workgroups (one per WGP).
// WG g owns h columns [32g,32g+32); its W_h slice lives in LDS for the
// whole kernel (loaded once). h planes ping-pong in global (L2-resident,
// bf16). Per step: 16 waves x 16 k-chunks x 1 WMMA, then tanh epilogue,
// then a cluster barrier (s_barrier_signal/-wait -3, NOP if unclustered)
// backed by a correct L2 atomic barrier.
// =====================================================================
__global__ __launch_bounds__(512)
void rnn_scan(const float* __restrict__ xp, const __bf16* __restrict__ Wh_img,
              const float* __restrict__ W_out, const float* __restrict__ b_out,
              __bf16* __restrict__ hbuf, unsigned* __restrict__ cnt,
              float* __restrict__ out) {
  __shared__ __bf16 Wsl[32][512];              // W_h[:, 32g:32g+32] as [nl][k], 32KB

  const int wg   = blockIdx.x;                 // 0..15
  const int tid  = threadIdx.x;
  const int lane = tid & 31;
  const int wv   = tid >> 5;                   // 16 waves
  const int mt   = wv & 7;                     // M-tile 0..7
  const int ntl  = wv >> 3;                    // local N-tile 0..1
  const int col  = lane & 15;
  const int hh   = lane >> 4;
  const int nl   = ntl * 16 + col;             // local column 0..31
  const int n    = wg * 32 + nl;               // global h column

  // --- one-time W_h slice load: contiguous 32KB (async-to-LDS path) ---
  {
    const __bf16* src = Wh_img + (long)wg * 32 * Hd;
    __bf16* dst = &Wsl[0][0];
    for (int i = 0; i < 4; ++i) {
      const int off = (i * 512 + tid) * 8;
      copy16_to_lds(dst + off, src + off);
    }
    async_copies_done();
  }
  __syncthreads();

  for (int t = 0; t < Tv; ++t) {
    // publish h_t (or h_0 from prep) to all 16 WGPs, then sync
#if HAVE_CLUSTER_BARRIER
    if (wv == 0) __builtin_amdgcn_s_cluster_barrier();   // fast path if clustered
#endif
    grid_barrier(cnt, (unsigned)(NWG * (t + 1)), tid);

    const __bf16* hc  = hbuf + (size_t)(t & 1) * (Bv * Hd);
    __bf16*       hnx = hbuf + (size_t)((t & 1) ^ 1) * (Bv * Hd);
    const float*  xpt = xp + (size_t)t * Bv * Hd;
    if (t + 1 < Tv)                            // pull next xp plane into L2/WGP$
      __builtin_prefetch(xp + (size_t)(t + 1) * Bv * Hd + tid * 128, 0, 3);

    v8f acc;
    for (int j = 0; j < 8; ++j) acc[j] = 0.0f;
    for (int kc = 0; kc < 16; ++kc) {          // K = 512
      const int k0 = kc * 32;
      const v16bf bfr = load_frag(&Wsl[nl][k0 + hh * 16],
                                  &Wsl[nl][k0 + hh * 16 + 8]);
      const __bf16* hrow = hc + (size_t)(mt * 16 + col) * Hd + k0;
      const v16bf af = load_frag(hrow + hh * 16, hrow + hh * 16 + 8);
      acc = __builtin_amdgcn_wmma_f32_16x16x32_bf16(
          false, af, false, bfr, (short)0, acc, false, false);
    }
    // h_{t+1}[m][n] = tanh(acc + xp_t[m][n]) for this wave's 16x16 tile
    for (int j = 0; j < 8; ++j) {
      const int m = mt * 16 + hh * 8 + j;
      const float val = acc[j] + xpt[(size_t)m * Hd + n];
      hnx[(size_t)m * Hd + n] = (__bf16)tanhf(val);
    }
  }

  // final publish of h_T (it lands in plane 0 since Tv is even)
#if HAVE_CLUSTER_BARRIER
  if (wv == 0) __builtin_amdgcn_s_cluster_barrier();
#endif
  grid_barrier(cnt, (unsigned)(NWG * (Tv + 1)), tid);

  // --- out = h_T @ W_out + b_out  (128 x 4), one WG ---
  if (wg == 0) {
    const __bf16* hT = hbuf;                   // plane 0
    const int b = tid >> 2, o = tid & 3;       // 512 threads = all outputs
    float s = b_out[o];
    for (int k = 0; k < Hd; ++k)
      s += (float)hT[(size_t)b * Hd + k] * W_out[k * 4 + o];
    out[b * 4 + o] = s;
  }
}

// =====================================================================
extern "C" void kernel_launch(void* const* d_in, const int* in_sizes, int n_in,
                              void* d_out, int out_size, void* d_ws, size_t ws_size,
                              hipStream_t stream) {
  const int*   x     = (const int*)  d_in[0];
  const float* embed = (const float*)d_in[1];
  const float* W_fc  = (const float*)d_in[2];
  const float* b_fc  = (const float*)d_in[3];
  const float* W_out = (const float*)d_in[4];
  const float* b_out = (const float*)d_in[5];

  // workspace layout
  char* base = (char*)d_ws;
  float*  xp     = (float*)base;                               // 512 MB
  const size_t XP_BYTES = (size_t)Tv * Bv * Hd * 4;
  __bf16* Wx_img = (__bf16*)(base + XP_BYTES);                 // 512 KB
  __bf16* Wh_img = Wx_img + (size_t)Hd * Hd;                   // 512 KB
  __bf16* hbuf   = Wh_img + (size_t)Hd * Hd;                   // 256 KB (2 planes)
  unsigned* cnt  = (unsigned*)(hbuf + (size_t)2 * Bv * Hd);    // 4 B

  prep    <<<(Hd * Hd) / 256, 256, 0, stream>>>(W_fc, Wx_img, Wh_img, hbuf, cnt);
  xp_gemm <<<(Bv * Tv) / 64,  256, 0, stream>>>(x, embed, Wx_img, b_fc, xp);
  rnn_scan<<<NWG, 512, 0, stream>>>(xp, Wh_img, W_out, b_out, hbuf, cnt,
                                    (float*)d_out);
}
